// NWnnSDP_25116968747015
// MI455X (gfx1250) — matrix operand order repaired and checked by reading, not compiled, not run on hardware
//
#include <hip/hip_runtime.h>
#include <math.h>

typedef __attribute__((ext_vector_type(16))) _Float16 v16h;
typedef __attribute__((ext_vector_type(8)))  _Float16 v8h;
typedef __attribute__((ext_vector_type(8)))  float    v8f;

namespace {

constexpr int BM  = 128;    // rows per block tile
constexpr int BN  = 128;    // cols per block tile
constexpr int BK  = 32;     // k per stage == WMMA K
constexpr int LDT = BK + 8; // padded LDS pitch (halfs): 80B rows, bank-friendly

enum {
  MODE_RELU   = 0,  // Y_f16[r][o] = relu(acc + bias[o])
  MODE_MUL    = 1,  // Y_f16[r][o] = (acc + bias[o]) * aux[r][o]
  MODE_MUL_T  = 2,  // Y_f16[o][r] = (acc + bias[o]) * aux[r][o]   (transposed store)
  MODE_EXP_T  = 3,  // Y_f16[o][r] = exp(-0.5*(acc*scale)^2)       (transposed store)
  MODE_SCALE  = 4,  // Y_f16[r][o] = acc * aux[r]                  (fold 1/denom)
  MODE_BIAS32 = 5,  // Y_f32[r][o] = acc + bias[o]
};

// Async DMA: global (16B/lane) -> LDS, tracked by ASYNCcnt.  ldsOff is the
// wave-relative LDS byte offset = low 32 bits of the generic shared pointer
// (generic = {SHARED_BASE aperture, as3 offset} per ISA flat-aperture rules).
__device__ __forceinline__ void async_g2l_b128(unsigned ldsOff, const void* gaddr) {
  asm volatile("global_load_async_to_lds_b128 %0, %1, off"
               :: "v"(ldsOff), "v"(gaddr) : "memory");
}

// A fragment, 16x32 f16 (ISA 7.12.2): M = lane%16; lanes<16 hold K 0-7 & 16-23,
// lanes>=16 hold K 8-15 & 24-31; VGPR j holds K pair (2j,2j+1) within its group.
__device__ __forceinline__ v16h frag_a(const _Float16* s, int row0, int lane) {
  const _Float16* p = s + (row0 + (lane & 15)) * LDT + ((lane & 16) ? 8 : 0);
  v16h a;
#pragma unroll
  for (int j = 0; j < 4; ++j) {
    a[2 * j]         = p[2 * j];
    a[2 * j + 1]     = p[2 * j + 1];
    a[8 + 2 * j]     = p[16 + 2 * j];
    a[8 + 2 * j + 1] = p[16 + 2 * j + 1];
  }
  return a;
}

// B fragment, 32x16: B[k][n] = Wtile[n][k] (row-major [col][k] tile).
// N = lane%16; lanes<16 hold K 0-7 & 16-23, lanes>=16 hold K 8-15 & 24-31, pairs per VGPR.
__device__ __forceinline__ v16h frag_b(const _Float16* s, int col0, int lane) {
  const _Float16* p = s + (col0 + (lane & 15)) * LDT + ((lane & 16) ? 8 : 0);
  v16h b;
#pragma unroll
  for (int j = 0; j < 16; ++j) b[j] = p[j < 8 ? j : j + 8];
  return b;
}

// Generic f16-WMMA GEMM:  acc[r][o] = sum_k X[r][k] * W[o][k], epilogue by MODE.
// Double-buffered LDS fed by async global->LDS DMA; copy of tile k+1 overlaps
// the 8 WMMAs of tile k.  Grid: (O/BN, R/BM, nbatch); dims are tile multiples.
template <int MODE>
__global__ __launch_bounds__(256)
void gemm16(const _Float16* __restrict__ X, long long xStride,
            const _Float16* __restrict__ W, long long wStride,
            const float* __restrict__ bias,
            const float* __restrict__ aux, long long auxStride, int ldAux,
            void* __restrict__ Yv, long long yStride,
            int K, int ldy, float scale)
{
  const int z = blockIdx.z;
  X += (size_t)z * xStride;
  W += (size_t)z * wStride;
  if (aux) aux += (size_t)z * auxStride;

  __shared__ _Float16 As[2][BM * LDT];
  __shared__ _Float16 Bs[2][BN * LDT];

  const int t    = threadIdx.x;
  const int lane = t & 31;
  const int wave = t >> 5;
  const int wr   = (wave >> 2) * 64;   // wave row origin in tile (2 wave-rows)
  const int wc   = (wave & 3) * 32;    // wave col origin in tile (4 wave-cols)

  const int ldRow = t >> 2;            // 0..63
  const int ldCol = (t & 3) << 3;      // 0,8,16,24 (8 halfs = b128)

  const size_t xrow = (size_t)blockIdx.y * BM + ldRow;
  const size_t wrow = (size_t)blockIdx.x * BN + ldRow;

  // Issue one stage (4 async b128 per wave -> ASYNCcnt += 4).
  auto stage = [&](int k0, int buf) {
    const _Float16* xs = X + xrow * (size_t)K + (k0 + ldCol);
    const unsigned  la = (unsigned)(size_t)&As[buf][ldRow * LDT + ldCol];
    async_g2l_b128(la,                        xs);
    async_g2l_b128(la + 64u * LDT * 2u,       xs + (size_t)64 * K);
    const _Float16* wsv = W + wrow * (size_t)K + (k0 + ldCol);
    const unsigned  lb  = (unsigned)(size_t)&Bs[buf][ldRow * LDT + ldCol];
    async_g2l_b128(lb,                        wsv);
    async_g2l_b128(lb + 64u * LDT * 2u,       wsv + (size_t)64 * K);
  };

  v8f c[4][2] = {};

  const int nIter = K / BK;
  stage(0, 0);
  for (int it = 0; it < nIter; ++it) {
    const int buf = it & 1;
    if (it > 0) __syncthreads();           // all waves done reading buf^1 (it-1)
    if (it + 1 < nIter) {
      stage((it + 1) * BK, buf ^ 1);       // prefetch next tile into other buffer
      asm volatile("s_wait_asynccnt %0" :: "i"(4));  // 4 oldest (= this buf) done
    } else {
      asm volatile("s_wait_asynccnt %0" :: "i"(0));
    }
    __syncthreads();                       // current buffer visible to all waves

    v16h a[4], b[2];
#pragma unroll
    for (int i = 0; i < 4; ++i) a[i] = frag_a(&As[buf][0], wr + 16 * i, lane);
#pragma unroll
    for (int j = 0; j < 2; ++j) b[j] = frag_b(&Bs[buf][0], wc + 16 * j, lane);
#pragma unroll
    for (int i = 0; i < 4; ++i)
#pragma unroll
      for (int j = 0; j < 2; ++j)
        c[i][j] = __builtin_amdgcn_wmma_f32_16x16x32_f16(
            false, a[i], false, b[j], (short)0, c[i][j], false, false);
  }

  // C/D layout: col = lane&15 (+16j), row = vgpr + 8*(lane>=16) (+16i)
  const int rTile = blockIdx.y * BM + wr + ((lane & 16) ? 8 : 0);
  const int oTile = blockIdx.x * BN + wc + (lane & 15);

#pragma unroll
  for (int i = 0; i < 4; ++i) {
#pragma unroll
    for (int j = 0; j < 2; ++j) {
      const v8f acc = c[i][j];
      const int r0 = rTile + 16 * i;
      const int o  = oTile + 16 * j;
      if (MODE == MODE_RELU) {
        _Float16* Y = (_Float16*)Yv + (size_t)z * yStride;
        const float bb = bias[o];
#pragma unroll
        for (int v = 0; v < 8; ++v)
          Y[(size_t)(r0 + v) * ldy + o] = (_Float16)fmaxf(acc[v] + bb, 0.0f);
      } else if (MODE == MODE_MUL) {
        _Float16* Y = (_Float16*)Yv + (size_t)z * yStride;
        const float bb = bias[o];
#pragma unroll
        for (int v = 0; v < 8; ++v)
          Y[(size_t)(r0 + v) * ldy + o] =
              (_Float16)((acc[v] + bb) * aux[(size_t)(r0 + v) * ldAux + o]);
      } else if (MODE == MODE_MUL_T) {
        _Float16* Y = (_Float16*)Yv + (size_t)z * yStride;
        const float bb = bias[o];
        v8h pk;
#pragma unroll
        for (int v = 0; v < 8; ++v)
          pk[v] = (_Float16)((acc[v] + bb) * aux[(size_t)(r0 + v) * ldAux + o]);
        *(v8h*)&Y[(size_t)o * ldy + r0] = pk;   // contiguous 8 halfs along r
      } else if (MODE == MODE_EXP_T) {
        _Float16* Y = (_Float16*)Yv + (size_t)z * yStride;
        v8h pk;
#pragma unroll
        for (int v = 0; v < 8; ++v) {
          const float d = acc[v] * scale;      // dist = S/sqrt(dk)
          pk[v] = (_Float16)__expf(-0.5f * d * d);  // logits <= 0: safe w/o max-sub
        }
        *(v8h*)&Y[(size_t)o * ldy + r0] = pk;
      } else if (MODE == MODE_SCALE) {
        _Float16* Y = (_Float16*)Yv + (size_t)z * yStride;
#pragma unroll
        for (int v = 0; v < 8; ++v)
          Y[(size_t)(r0 + v) * ldy + o] = (_Float16)(acc[v] * aux[r0 + v]);
      } else { // MODE_BIAS32
        float* Y = (float*)Yv + (size_t)z * yStride;
        const float bb = bias[o];
#pragma unroll
        for (int v = 0; v < 8; ++v)
          Y[(size_t)(r0 + v) * ldy + o] = acc[v] + bb;
      }
    }
  }
}

__global__ void cvt_f16(const float* __restrict__ x, _Float16* __restrict__ y, int n) {
  int i = blockIdx.x * 256 + threadIdx.x;
  if (i < n) y[i] = (_Float16)x[i];
}

// invden[row] = 1 / sum_n P[row][n]  (deterministic block reduction, no atomics)
__global__ void rowsum_recip(const _Float16* __restrict__ P, float* __restrict__ invden,
                             int ncol) {
  const _Float16* row = P + (size_t)blockIdx.x * ncol;
  float s = 0.0f;
  for (int i = threadIdx.x; i < ncol; i += 256) s += (float)row[i];
  __shared__ float red[256];
  red[threadIdx.x] = s;
  __syncthreads();
  for (int off = 128; off > 0; off >>= 1) {
    if ((int)threadIdx.x < off) red[threadIdx.x] += red[threadIdx.x + off];
    __syncthreads();
  }
  if (threadIdx.x == 0) invden[blockIdx.x] = 1.0f / red[0];
}

} // namespace

extern "C" void kernel_launch(void* const* d_in, const int* in_sizes, int n_in,
                              void* d_out, int out_size, void* d_ws, size_t ws_size,
                              hipStream_t stream) {
  (void)in_sizes; (void)n_in; (void)out_size; (void)ws_size;
  constexpr int B = 2, N = 2048, M = 2048, DK = 512, H = 512;

  const float* KEY   = (const float*)d_in[0];
  const float* VALUE = (const float*)d_in[1];
  const float* QUERY = (const float*)d_in[2];
  const float* W1w = (const float*)d_in[3];  const float* W1b = (const float*)d_in[4];
  const float* W2w = (const float*)d_in[5];  const float* W2b = (const float*)d_in[6];
  const float* W3w = (const float*)d_in[7];  const float* W3b = (const float*)d_in[8];
  const float* V1w = (const float*)d_in[9];  const float* V1b = (const float*)d_in[10];
  const float* V2w = (const float*)d_in[11]; const float* V2b = (const float*)d_in[12];
  const float* V3w = (const float*)d_in[13]; const float* V3b = (const float*)d_in[14];
  const float* Wow = (const float*)d_in[15]; const float* Wob = (const float*)d_in[16];

  char* wp = (char*)d_ws;
  auto take = [&](size_t bytes) {
    char* p = wp; wp += (bytes + 255) & ~(size_t)255; return p;
  };
  _Float16* KEYh = (_Float16*)take((size_t)B * N * DK * 2);
  _Float16* QRYh = (_Float16*)take((size_t)B * M * DK * 2);
  _Float16* VALh = (_Float16*)take((size_t)B * N * H * 2);
  _Float16* W1h = (_Float16*)take((size_t)H * DK * 2);
  _Float16* W2h = (_Float16*)take((size_t)H * H * 2);
  _Float16* W3h = (_Float16*)take((size_t)DK * H * 2);
  _Float16* V1h = (_Float16*)take((size_t)H * H * 2);
  _Float16* V2h = (_Float16*)take((size_t)H * H * 2);
  _Float16* V3h = (_Float16*)take((size_t)H * H * 2);
  _Float16* Woh = (_Float16*)take((size_t)H * H * 2);
  _Float16* h1  = (_Float16*)take((size_t)B * N * H * 2);   // reused per branch
  _Float16* h2  = (_Float16*)take((size_t)B * N * H * 2);
  _Float16* Ks  = (_Float16*)take((size_t)B * N * DK * 2);
  _Float16* Qs  = (_Float16*)take((size_t)B * M * DK * 2);
  _Float16* VsT = (_Float16*)take((size_t)B * H * N * 2);   // [b][h][n]
  _Float16* Pt  = (_Float16*)take((size_t)B * M * N * 2);   // [b][m][n] = P^T
  float*    invden = (float*)take((size_t)B * M * 4);
  _Float16* ctxn = (_Float16*)take((size_t)B * M * H * 2);  // normalized ctx, f16

  auto cvt = [&](const float* s, _Float16* d, int n) {
    cvt_f16<<<dim3((n + 255) / 256), dim3(256), 0, stream>>>(s, d, n);
  };
  cvt(KEY, KEYh, B * N * DK);
  cvt(QUERY, QRYh, B * M * DK);
  cvt(VALUE, VALh, B * N * H);
  cvt(W1w, W1h, H * DK); cvt(W2w, W2h, H * H); cvt(W3w, W3h, DK * H);
  cvt(V1w, V1h, H * H);  cvt(V2w, V2h, H * H); cvt(V3w, V3h, H * H);
  cvt(Wow, Woh, H * H);

  const dim3 blk(256);
  const long long LL0 = 0;

  // ---- W-branch MLP on KEY -> Ks (gated) ----
  gemm16<MODE_RELU><<<dim3(H / BN, (B * N) / BM, 1), blk, 0, stream>>>(
      KEYh, LL0, W1h, LL0, W1b, nullptr, LL0, 0, h1, LL0, DK, H, 1.0f);
  gemm16<MODE_RELU><<<dim3(H / BN, (B * N) / BM, 1), blk, 0, stream>>>(
      h1, LL0, W2h, LL0, W2b, nullptr, LL0, 0, h2, LL0, H, H, 1.0f);
  gemm16<MODE_MUL><<<dim3(DK / BN, (B * N) / BM, 1), blk, 0, stream>>>(
      h2, LL0, W3h, LL0, W3b, KEY, LL0, DK, Ks, LL0, H, DK, 1.0f);

  // ---- W-branch MLP on QUERY -> Qs (gated) ----
  gemm16<MODE_RELU><<<dim3(H / BN, (B * M) / BM, 1), blk, 0, stream>>>(
      QRYh, LL0, W1h, LL0, W1b, nullptr, LL0, 0, h1, LL0, DK, H, 1.0f);
  gemm16<MODE_RELU><<<dim3(H / BN, (B * M) / BM, 1), blk, 0, stream>>>(
      h1, LL0, W2h, LL0, W2b, nullptr, LL0, 0, h2, LL0, H, H, 1.0f);
  gemm16<MODE_MUL><<<dim3(DK / BN, (B * M) / BM, 1), blk, 0, stream>>>(
      h2, LL0, W3h, LL0, W3b, QUERY, LL0, DK, Qs, LL0, H, DK, 1.0f);

  // ---- V-branch MLP on VALUE -> VsT (gated, transposed per batch) ----
  gemm16<MODE_RELU><<<dim3(H / BN, (B * N) / BM, 1), blk, 0, stream>>>(
      VALh, LL0, V1h, LL0, V1b, nullptr, LL0, 0, h1, LL0, H, H, 1.0f);
  gemm16<MODE_RELU><<<dim3(H / BN, (B * N) / BM, 1), blk, 0, stream>>>(
      h1, LL0, V2h, LL0, V2b, nullptr, LL0, 0, h2, LL0, H, H, 1.0f);
  gemm16<MODE_MUL_T><<<dim3(H / BN, N / BM, B), blk, 0, stream>>>(
      h2, (long long)N * H, V3h, LL0, V3b, VALUE, (long long)N * H, H,
      VsT, (long long)H * N, H, /*ldy=N*/ N, 1.0f);

  // ---- attention probabilities: Pt[b][m][n] = exp(-0.5*(S/sqrt(dk))^2) ----
  const float sc = 1.0f / sqrtf((float)DK);
  gemm16<MODE_EXP_T><<<dim3(M / BN, N / BM, B), blk, 0, stream>>>(
      Ks, (long long)N * DK, Qs, (long long)M * DK, nullptr, nullptr, LL0, 0,
      Pt, (long long)M * N, DK, /*ldy=N*/ N, sc);

  // ---- softmax denominators (per (b,m) row of Pt), deterministic ----
  rowsum_recip<<<dim3(B * M), dim3(256), 0, stream>>>(Pt, invden, N);

  // ---- context (normalization folded): ctxn[b][m][h] = invden[m] * Pt[m][:] @ VsT[h][:] ----
  gemm16<MODE_SCALE><<<dim3(H / BN, M / BM, B), blk, 0, stream>>>(
      Pt, (long long)M * N, VsT, (long long)H * N, nullptr, invden, (long long)M, 0,
      ctxn, (long long)M * H, N, H, 1.0f);

  // ---- output projection -> d_out (f32) ----
  gemm16<MODE_BIAS32><<<dim3(H / BN, (B * M) / BM, 1), blk, 0, stream>>>(
      ctxn, LL0, Woh, LL0, Wob, nullptr, LL0, 0, d_out, LL0, H, H, 1.0f);
}
